// ConditionedPNA_54984171323691
// MI455X (gfx1250) — matrix operand order repaired and checked by reading, not compiled, not run on hardware
//
#include <hip/hip_runtime.h>
#include <math.h>

typedef __bf16 bf16_t;
typedef __attribute__((ext_vector_type(16))) __bf16 v16bf;
typedef __attribute__((ext_vector_type(8)))  __bf16 v8bf;
typedef __attribute__((ext_vector_type(4)))  __bf16 v4bf;
typedef __attribute__((ext_vector_type(8)))  float  v8f;
typedef __attribute__((ext_vector_type(4)))  float  v4f;

#define DF 128
#define NAGG 4

// ----------------- fragment loaders (CDNA5 WMMA 16-bit layouts) -----------------
// A 16x32 bf16: lanes 0-15 = rows M0..15, elems 0..7 = K k0+half*8.., elems 8..15 = K k0+16+half*8..
__device__ __forceinline__ v16bf load_fragA(const bf16_t* base, int m0, int ldk,
                                            int k0, int lane, int M) {
  int hf = lane >> 4, r = lane & 15;
  int row = m0 + r; if (row >= M) row = M - 1;
  const bf16_t* p = base + (size_t)row * ldk;
  v8bf lo = *(const v8bf*)(p + k0 + hf * 8);
  v8bf hi = *(const v8bf*)(p + k0 + 16 + hf * 8);
  v16bf a;
#pragma unroll
  for (int i = 0; i < 8; ++i) { a[i] = lo[i]; a[8 + i] = hi[i]; }
  return a;
}

// B 32x16 bf16 from transposed weights WT[n][k]: lane = col, elems 0..7 = K k0+half*8.., elems 8..15 = K k0+16+half*8..
__device__ __forceinline__ v16bf load_fragB(const bf16_t* WT, int n0, int ldk,
                                            int k0, int lane) {
  int hf = lane >> 4, r = lane & 15;
  const bf16_t* p = WT + (size_t)(n0 + r) * ldk + k0 + hf * 8;
  v8bf lo = *(const v8bf*)(p);
  v8bf hi = *(const v8bf*)(p + 16);
  v16bf b;
#pragma unroll
  for (int i = 0; i < 8; ++i) { b[i] = lo[i]; b[8 + i] = hi[i]; }
  return b;
}

// ----------------- generic bf16 GEMM: C[M x 128] = A[M x K] @ WT^T (+bias)(+residual) ------
__global__ __launch_bounds__(256) void gemm_bf16_kernel(
    const bf16_t* __restrict__ A, const bf16_t* __restrict__ WT,
    const float* __restrict__ bias, const float* __restrict__ residual,
    float* __restrict__ C, int M, int K) {
  int lane = threadIdx.x & 31;
  int wave = threadIdx.x >> 5;
  int m0 = blockIdx.x * 16;
  int n0 = wave * 16;
  v8f acc = {};
  for (int k0 = 0; k0 < K; k0 += 32) {
    v16bf a = load_fragA(A, m0, K, k0, lane, M);
    v16bf b = load_fragB(WT, n0, K, k0, lane);
    acc = __builtin_amdgcn_wmma_f32_16x16x32_bf16(false, a, false, b, (short)0,
                                                  acc, false, false);
  }
  int hf = lane >> 4, r = lane & 15;
  int col = n0 + r;
#pragma unroll
  for (int v = 0; v < 8; ++v) {
    int row = m0 + v + 8 * hf;
    if (row < M) {
      float val = acc[v];
      if (bias)     val += bias[col];
      if (residual) val += residual[(size_t)row * DF + col];
      C[(size_t)row * DF + col] = val;
    }
  }
}

// ----------------- post GEMM: A assembled on the fly from [h | agg | agg*amp | agg*att] ----
__device__ __forceinline__ void load_chunk_post(
    const bf16_t* hb, const bf16_t* aggb, float ampv, float attv,
    int row, int cb, bf16_t* out8) {
  int j  = cb >> 7;   // 13-block index
  int kk = cb & 127;
  const bf16_t* src; float scale = 1.f;
  if (j == 0) {
    src = hb + (size_t)row * DF + kk;
  } else {
    int g = (j - 1) >> 2, ai = (j - 1) & 3;
    src = aggb + (size_t)row * (NAGG * DF) + ai * DF + kk;
    scale = (g == 0) ? 1.f : (g == 1 ? ampv : attv);
  }
  v8bf v = *(const v8bf*)src;
#pragma unroll
  for (int i = 0; i < 8; ++i) out8[i] = (bf16_t)((float)v[i] * scale);
}

__global__ __launch_bounds__(256) void gemm_post_kernel(
    const bf16_t* __restrict__ hb, const bf16_t* __restrict__ aggb,
    const float* __restrict__ amp, const float* __restrict__ att,
    const bf16_t* __restrict__ WT, const float* __restrict__ bias,
    float* __restrict__ C, int M) {
  const int K = (1 + 3 * NAGG) * DF;  // 1664
  int lane = threadIdx.x & 31;
  int wave = threadIdx.x >> 5;
  int m0 = blockIdx.x * 16;
  int n0 = wave * 16;
  int hf = lane >> 4, r = lane & 15;
  int row = m0 + r; if (row >= M) row = M - 1;
  float ampv = amp[row], attv = att[row];
  v8f acc = {};
  for (int k0 = 0; k0 < K; k0 += 32) {
    bf16_t t0[8], t1[8];
    load_chunk_post(hb, aggb, ampv, attv, row, k0 + hf * 8, t0);
    load_chunk_post(hb, aggb, ampv, attv, row, k0 + 16 + hf * 8, t1);
    v16bf a;
#pragma unroll
    for (int i = 0; i < 8; ++i) { a[i] = t0[i]; a[8 + i] = t1[i]; }
    v16bf b = load_fragB(WT, n0, K, k0, lane);
    acc = __builtin_amdgcn_wmma_f32_16x16x32_bf16(false, a, false, b, (short)0,
                                                  acc, false, false);
  }
  int col = n0 + r;
#pragma unroll
  for (int v = 0; v < 8; ++v) {
    int rowv = m0 + v + 8 * hf;
    if (rowv < M) C[(size_t)rowv * DF + col] = acc[v] + bias[col];
  }
}

// ----------------- graph preprocessing ----------------------------------------------------
__global__ void hist_kernel(const int* __restrict__ dst, int* __restrict__ cnt, int E) {
  for (int i = blockIdx.x * blockDim.x + threadIdx.x; i < E; i += gridDim.x * blockDim.x)
    atomicAdd(&cnt[dst[i]], 1);
}

__global__ void scan_kernel(const int* __restrict__ cnt, int* __restrict__ rowptr,
                            int* __restrict__ wrk, int n) {
  __shared__ int sdata[256];
  __shared__ int running;
  if (threadIdx.x == 0) running = 0;
  __syncthreads();
  for (int base = 0; base < n; base += 256) {
    int i = base + (int)threadIdx.x;
    int v = (i < n) ? cnt[i] : 0;
    sdata[threadIdx.x] = v;
    __syncthreads();
    for (int off = 1; off < 256; off <<= 1) {
      int t = (threadIdx.x >= (unsigned)off) ? sdata[threadIdx.x - off] : 0;
      __syncthreads();
      sdata[threadIdx.x] += t;
      __syncthreads();
    }
    int excl = sdata[threadIdx.x] - v;
    if (i < n) { rowptr[i] = running + excl; wrk[i] = running + excl; }
    __syncthreads();
    if (threadIdx.x == 255) running += sdata[255];
    __syncthreads();
  }
  if (threadIdx.x == 0) rowptr[n] = running;
}

__global__ void scatter_kernel(const int* __restrict__ src, const int* __restrict__ dst,
                               int* __restrict__ wrk, int* __restrict__ colidx, int E) {
  for (int i = blockIdx.x * blockDim.x + threadIdx.x; i < E; i += gridDim.x * blockDim.x) {
    int p = atomicAdd(&wrk[dst[i]], 1);
    colidx[p] = src[i];
  }
}

__global__ void logd_partial_kernel(const int* __restrict__ cnt, float* __restrict__ logd,
                                    float* __restrict__ partial, int n) {
  __shared__ float sdata[256];
  float local = 0.f;
  for (int i = blockIdx.x * 256 + threadIdx.x; i < n; i += gridDim.x * 256) {
    float c = (float)cnt[i];
    logd[i] = logf(fmaxf(c, 1.f) + 1.f);
    local += logf(c + 1.f);
  }
  sdata[threadIdx.x] = local;
  __syncthreads();
  for (int off = 128; off > 0; off >>= 1) {
    if (threadIdx.x < (unsigned)off) sdata[threadIdx.x] += sdata[threadIdx.x + off];
    __syncthreads();
  }
  if (threadIdx.x == 0) partial[blockIdx.x] = sdata[0];
}

__global__ void finalize_avg_kernel(const float* __restrict__ partial, float* __restrict__ avg,
                                    int nb, float invN) {
  if (threadIdx.x == 0 && blockIdx.x == 0) {
    float s = 0.f;
    for (int i = 0; i < nb; ++i) s += partial[i];  // deterministic order
    *avg = s * invN;
  }
}

// ----------------- per-node segment reduction (CSR; one wave per node, float4 per lane) ---
__global__ __launch_bounds__(256) void aggregate_kernel(
    const float* __restrict__ A, const float* __restrict__ B2,
    const int* __restrict__ rowptr, const int* __restrict__ colidx,
    const int* __restrict__ cnt, const float* __restrict__ logd,
    const float* __restrict__ avgp, bf16_t* __restrict__ aggb,
    float* __restrict__ amp, float* __restrict__ att, int n) {
  int wave = threadIdx.x >> 5;
  int lane = threadIdx.x & 31;
  int i = blockIdx.x * 8 + wave;          // 8 nodes per 256-thread block
  if (i >= n) return;
  int beg = rowptr[i], end = rowptr[i + 1];
  v4f s = {0.f, 0.f, 0.f, 0.f};
  v4f ss = {0.f, 0.f, 0.f, 0.f};
  v4f mx = {-3.402823466e38f, -3.402823466e38f, -3.402823466e38f, -3.402823466e38f};
  v4f mn = {3.402823466e38f, 3.402823466e38f, 3.402823466e38f, 3.402823466e38f};
  const int fo = lane * 4;                // 4 consecutive features per lane
  for (int e = beg; e < end; ++e) {
    int j = colidx[e];
    if (e + 4 < end) __builtin_prefetch(&B2[(size_t)colidx[e + 4] * DF + fo], 0, 1);
    v4f v = *(const v4f*)(B2 + (size_t)j * DF + fo);
#pragma unroll
    for (int q = 0; q < 4; ++q) {
      s[q] += v[q];
      ss[q] += v[q] * v[q];
      mx[q] = fmaxf(mx[q], v[q]);
      mn[q] = fminf(mn[q], v[q]);
    }
  }
  float c  = (float)cnt[i];
  float dn = fmaxf(c, 1.f);
  bool has = c > 0.f;
  v4f a = *(const v4f*)(A + (size_t)i * DF + fo);
  v4bf o_mean, o_mx, o_mn, o_sd;
#pragma unroll
  for (int q = 0; q < 4; ++q) {
    float mean   = (c * a[q] + s[q]) / dn;
    float meansq = (c * a[q] * a[q] + 2.f * a[q] * s[q] + ss[q]) / dn;
    float var    = meansq - mean * mean;
    float sd     = sqrtf(fmaxf(var, 0.f) + 1e-5f);
    o_mean[q] = (bf16_t)mean;
    o_mx[q]   = (bf16_t)(has ? (a[q] + mx[q]) : 0.f);
    o_mn[q]   = (bf16_t)(has ? (a[q] + mn[q]) : 0.f);
    o_sd[q]   = (bf16_t)sd;
  }
  size_t base = (size_t)i * (NAGG * DF);
  *(v4bf*)(aggb + base + 0 * DF + fo) = o_mean;
  *(v4bf*)(aggb + base + 1 * DF + fo) = o_mx;
  *(v4bf*)(aggb + base + 2 * DF + fo) = o_mn;
  *(v4bf*)(aggb + base + 3 * DF + fo) = o_sd;
  if (lane == 0) {
    float ld = logd[i], al = *avgp;
    amp[i] = ld / al;
    att[i] = al / ld;
  }
}

// ----------------- conversions -------------------------------------------------------------
__global__ void f2b_kernel(const float* __restrict__ src, bf16_t* __restrict__ dst, size_t n) {
  for (size_t i = blockIdx.x * 256ull + threadIdx.x; i < n; i += (size_t)gridDim.x * 256ull)
    dst[i] = (bf16_t)src[i];
}

// src f32 [K x 128] row-major -> dst bf16 [128 x K] (transposed, n-major)
__global__ void convert_wt_kernel(const float* __restrict__ src, bf16_t* __restrict__ dst, int K) {
  int idx = blockIdx.x * 256 + threadIdx.x;
  int total = K * DF;
  if (idx < total) {
    int k = idx / DF, n = idx % DF;
    dst[(size_t)n * K + k] = (bf16_t)src[idx];
  }
}

// ----------------- launch ------------------------------------------------------------------
extern "C" void kernel_launch(void* const* d_in, const int* in_sizes, int n_in,
                              void* d_out, int out_size, void* d_ws, size_t ws_size,
                              hipStream_t stream) {
  const float* x     = (const float*)d_in[0];
  const int*   eidx  = (const int*)  d_in[1];
  const float* preW  = (const float*)d_in[2];
  const float* preB  = (const float*)d_in[3];
  const float* postW = (const float*)d_in[4];
  const float* postB = (const float*)d_in[5];
  const float* linW  = (const float*)d_in[6];
  const float* linB  = (const float*)d_in[7];

  const int N = in_sizes[0] / DF;
  const int E = in_sizes[1] / 2;
  const int L = in_sizes[2] / (2 * DF * DF);
  const int KPOST = (1 + 3 * NAGG) * DF;  // 1664

  const int* src = eidx;
  const int* dst = eidx + E;

  // workspace carve-out (256B aligned)
  char* cur = (char*)d_ws;
  auto carve = [&](size_t bytes) -> void* {
    void* r = cur;
    cur += (bytes + 255) & ~(size_t)255;
    return r;
  };
  int*    cnt     = (int*)   carve((size_t)N * 4);
  int*    rowptr  = (int*)   carve((size_t)(N + 1) * 4);
  int*    wrk     = (int*)   carve((size_t)(N + 1) * 4);
  int*    colidx  = (int*)   carve((size_t)E * 4);
  float*  logd    = (float*) carve((size_t)N * 4);
  float*  partial = (float*) carve(256 * 4);
  float*  avgp    = (float*) carve(4);
  float*  hbuf0   = (float*) carve((size_t)N * DF * 4);
  float*  hbuf1   = (float*) carve((size_t)N * DF * 4);
  bf16_t* hb      = (bf16_t*)carve((size_t)N * DF * 2);
  float*  Amat    = (float*) carve((size_t)N * DF * 4);
  float*  Bmat    = (float*) carve((size_t)N * DF * 4);
  float*  O1      = (float*) carve((size_t)N * DF * 4);
  bf16_t* O1b     = (bf16_t*)carve((size_t)N * DF * 2);
  bf16_t* aggb    = (bf16_t*)carve((size_t)N * NAGG * DF * 2);
  float*  amp     = (float*) carve((size_t)N * 4);
  float*  att     = (float*) carve((size_t)N * 4);
  bf16_t* preW0t  = (bf16_t*)carve((size_t)L * DF * DF * 2);
  bf16_t* preW1t  = (bf16_t*)carve((size_t)L * DF * DF * 2);
  bf16_t* postWt  = (bf16_t*)carve((size_t)L * KPOST * DF * 2);
  bf16_t* linWt   = (bf16_t*)carve((size_t)L * DF * DF * 2);

  const int eb = (E + 255) / 256;
  const int mt = N / 16 + ((N % 16) ? 1 : 0);
  const size_t feat = (size_t)N * DF;
  const int fb = (int)((feat + 255) / 256);

  // ---- graph structure (CSR) + degree statistics ----
  hipMemsetAsync(cnt, 0, (size_t)N * 4, stream);
  hist_kernel<<<eb, 256, 0, stream>>>(dst, cnt, E);
  scan_kernel<<<1, 256, 0, stream>>>(cnt, rowptr, wrk, N);
  scatter_kernel<<<eb, 256, 0, stream>>>(src, dst, wrk, colidx, E);
  logd_partial_kernel<<<256, 256, 0, stream>>>(cnt, logd, partial, N);
  finalize_avg_kernel<<<1, 1, 0, stream>>>(partial, avgp, 256, 1.f / (float)N);

  // ---- weight conversion: transposed bf16 copies ----
  for (int l = 0; l < L; ++l) {
    convert_wt_kernel<<<(DF * DF + 255) / 256, 256, 0, stream>>>(
        preW + (size_t)l * 2 * DF * DF, preW0t + (size_t)l * DF * DF, DF);
    convert_wt_kernel<<<(DF * DF + 255) / 256, 256, 0, stream>>>(
        preW + (size_t)l * 2 * DF * DF + (size_t)DF * DF, preW1t + (size_t)l * DF * DF, DF);
    convert_wt_kernel<<<(KPOST * DF + 255) / 256, 256, 0, stream>>>(
        postW + (size_t)l * KPOST * DF, postWt + (size_t)l * KPOST * DF, KPOST);
    convert_wt_kernel<<<(DF * DF + 255) / 256, 256, 0, stream>>>(
        linW + (size_t)l * DF * DF, linWt + (size_t)l * DF * DF, DF);
  }

  // ---- layers (ping-pong h buffers; final layer writes d_out) ----
  const float* h_cur = x;
  for (int l = 0; l < L; ++l) {
    float* h_next = (l == L - 1) ? (float*)d_out : ((l == 0) ? hbuf0 : hbuf1);

    // h -> bf16
    f2b_kernel<<<fb, 256, 0, stream>>>(h_cur, hb, feat);

    // A = h@W0 + preb  (dst side, bias folded);  B = h@W1 (src side)
    gemm_bf16_kernel<<<mt, 256, 0, stream>>>(hb, preW0t + (size_t)l * DF * DF,
                                             preB + (size_t)l * DF, nullptr, Amat, N, DF);
    gemm_bf16_kernel<<<mt, 256, 0, stream>>>(hb, preW1t + (size_t)l * DF * DF,
                                             nullptr, nullptr, Bmat, N, DF);

    // per-node segment statistics over CSR neighbors
    aggregate_kernel<<<(N + 7) / 8, 256, 0, stream>>>(Amat, Bmat, rowptr, colidx, cnt, logd,
                                                      avgp, aggb, amp, att, N);

    // post_nn: [h | agg | agg*amp | agg*att] @ postW + postb
    gemm_post_kernel<<<mt, 256, 0, stream>>>(hb, aggb, amp, att,
                                             postWt + (size_t)l * KPOST * DF,
                                             postB + (size_t)l * DF, O1, N);

    // lin + residual
    f2b_kernel<<<fb, 256, 0, stream>>>(O1, O1b, feat);
    gemm_bf16_kernel<<<mt, 256, 0, stream>>>(O1b, linWt + (size_t)l * DF * DF,
                                             linB + (size_t)l * DF, h_cur, h_next, N, DF);

    h_cur = h_next;
  }
}
